// GAT_stat_14766097563855
// MI455X (gfx1250) — matrix operand order repaired and checked by reading, not compiled, not run on hardware
//
#include <hip/hip_runtime.h>
#include <hip/hip_bf16.h>
#include <math.h>

// Problem constants (match reference)
#define NN   50000
#define EE   800000
#define HH   4
#define CC   32
#define HC   128
#define LLAY 3
#define GG   64
#define OUTD 64
#define NSLP 0.2f
#define ETOT (EE + NN)

typedef __attribute__((ext_vector_type(16))) __bf16 v16bf;
typedef __attribute__((ext_vector_type(8)))  float  v8f;

// float atomic max via signed/unsigned int trick (correct for all values incl. -inf init)
__device__ __forceinline__ void atomicMaxF(float* addr, float v) {
  if (v >= 0.0f) atomicMax((int*)addr, __float_as_int(v));
  else           atomicMin((unsigned int*)addr, __float_as_uint(v));
}

// ---------------- utility kernels ----------------
__global__ void k_fill(float* p, float v, int n) {
  int i = blockIdx.x * blockDim.x + threadIdx.x;
  if (i < n) p[i] = v;
}

__global__ void k_counts(const int* __restrict__ batch, float* __restrict__ counts) {
  int i = blockIdx.x * blockDim.x + threadIdx.x;
  if (i < NN) atomicAdd(&counts[batch[i]], 1.0f);
}

__global__ void k_f32_to_bf16(const float* __restrict__ in, unsigned short* __restrict__ out, int n) {
  int i = blockIdx.x * blockDim.x + threadIdx.x;
  if (i < n) {
    __bf16 b = (__bf16)in[i];
    out[i] = __builtin_bit_cast(unsigned short, b);
  }
}

// Wt[j][k] = W[k][j], converted to bf16 (so GEMM B-loads are contiguous in K)
__global__ void k_w_transpose_bf16(const float* __restrict__ W, unsigned short* __restrict__ Wt) {
  int i = blockIdx.x * blockDim.x + threadIdx.x;
  if (i < HC * HC) {
    int k = i / HC, j = i - k * HC;
    __bf16 b = (__bf16)W[i];
    Wt[j * HC + k] = __builtin_bit_cast(unsigned short, b);
  }
}

// ---------------- WMMA GEMM: h = x @ W + bias  (x:[NN,HC] bf16, Wt:[HC,HC] bf16 transposed)
// One wave per 16x16 output tile; 8 waves/block cover the 8 column tiles of HC=128.
// K loop: 4 x v_wmma_f32_16x16x32_bf16 (f32 accumulate).
__global__ void k_gemm_wmma(const unsigned short* __restrict__ xb_us,
                            const unsigned short* __restrict__ wtb_us,
                            const float* __restrict__ bias,
                            float* __restrict__ hout) {
  const __bf16* A  = (const __bf16*)xb_us;   // [NN, HC] row-major
  const __bf16* Bt = (const __bf16*)wtb_us;  // [HC, HC], Bt[col][k] = W[k][col]
  const int wv   = threadIdx.x >> 5;         // col tile 0..7
  const int lane = threadIdx.x & 31;
  const int half = lane >> 4;
  const int l16  = lane & 15;
  const int row  = blockIdx.x * 16 + l16;    // A row held by this lane
  const int col  = wv * 16 + l16;            // B column held by this lane

  v8f acc = {};
#pragma unroll
  for (int kb = 0; kb < HC; kb += 32) {
    // A 16x32 bf16 layout: lane holds M=l16; K = kb+8*half+{0..7} and kb+16+8*half+{0..7}
    const __bf16* ap = A + (size_t)row * HC + kb + 8 * half;
    // B 32x16 bf16 layout: lane holds N=l16; K = kb+8*half+{0..7} and kb+16+8*half+{0..7}
    const __bf16* bp = Bt + (size_t)col * HC + kb + 8 * half;
    v16bf a, b;
#pragma unroll
    for (int j = 0; j < 8; ++j) { a[j] = ap[j]; a[8 + j] = ap[16 + j]; }
#pragma unroll
    for (int j = 0; j < 8; ++j) { b[j] = bp[j]; b[8 + j] = bp[16 + j]; }
    acc = __builtin_amdgcn_wmma_f32_16x16x32_bf16(false, a, false, b, (short)0, acc, false, false);
  }
  // C/D layout: VGPR v -> M = v + 8*half, N = l16
  const float bc = bias[col];
#pragma unroll
  for (int v = 0; v < 8; ++v) {
    int r = blockIdx.x * 16 + v + 8 * half;
    hout[(size_t)r * HC + col] = acc[v] + bc;
  }
}

// ---------------- attention logits: al_s/al_d = sum_c h[n,h,c]*att[h,c] ----------------
__global__ void k_attn_logits(const float* __restrict__ h, const float* __restrict__ asrc,
                              const float* __restrict__ adst, float* __restrict__ al_s,
                              float* __restrict__ al_d) {
  int t = blockIdx.x * blockDim.x + threadIdx.x; // over N*H
  if (t >= NN * HH) return;
  int hd = t & (HH - 1);
  const float* hp = h + (size_t)t * CC;          // [N,H,C] flat == [N*H, C]
  const float* as = asrc + hd * CC;
  const float* ad = adst + hd * CC;
  float ss = 0.f, sd = 0.f;
#pragma unroll
  for (int c = 0; c < CC; ++c) { float v = hp[c]; ss += v * as[c]; sd += v * ad[c]; }
  al_s[t] = ss; al_d[t] = sd;
}

// ---------------- edge stage (E real edges + N self loops) ----------------
__global__ void k_edge_max(const int* __restrict__ src, const int* __restrict__ dst,
                           const float* __restrict__ als, const float* __restrict__ ald,
                           float* __restrict__ earr, float* __restrict__ mmax) {
  int e = blockIdx.x * blockDim.x + threadIdx.x;
  if (e >= ETOT) return;
  int s, d;
  if (e < EE) { s = src[e]; d = dst[e]; } else { s = e - EE; d = s; }
#pragma unroll
  for (int h = 0; h < HH; ++h) {
    float v = als[s * HH + h] + ald[d * HH + h];
    v = v > 0.f ? v : NSLP * v;               // leaky relu
    earr[(size_t)e * HH + h] = v;
    atomicMaxF(&mmax[d * HH + h], v);
  }
}

__global__ void k_edge_expsum(const int* __restrict__ dst, float* __restrict__ earr,
                              const float* __restrict__ mmax, float* __restrict__ ssum) {
  int e = blockIdx.x * blockDim.x + threadIdx.x;
  if (e >= ETOT) return;
  int d = (e < EE) ? dst[e] : e - EE;
#pragma unroll
  for (int h = 0; h < HH; ++h) {
    float ex = __expf(earr[(size_t)e * HH + h] - mmax[d * HH + h]);
    earr[(size_t)e * HH + h] = ex;
    atomicAdd(&ssum[d * HH + h], ex);
  }
}

__global__ void k_edge_scatter(const int* __restrict__ src, const int* __restrict__ dst,
                               const float* __restrict__ earr, const float* __restrict__ ssum,
                               const float* __restrict__ hfeat, float* __restrict__ xnext) {
  int t = blockIdx.x * blockDim.x + threadIdx.x; // over ETOT*HC (108.8M, fits int)
  if (t >= ETOT * HC) return;
  int f = t & (HC - 1);
  int e = t >> 7;
  int s, d;
  if (e < EE) { s = src[e]; d = dst[e]; } else { s = e - EE; d = s; }
  int h = f >> 5;                                // C=32
  float alpha = earr[(size_t)e * HH + h] / (ssum[d * HH + h] + 1e-16f);
  atomicAdd(&xnext[(size_t)d * HC + f], hfeat[(size_t)s * HC + f] * alpha);
}

__global__ void k_relu(float* x, int n) {
  int i = blockIdx.x * blockDim.x + threadIdx.x;
  if (i < n) x[i] = fmaxf(x[i], 0.f);
}

// ---------------- pooling ----------------
// gate[n] = sigmoid(x[n]·gw + gb); atomic max of gate per graph. One wave per node.
__global__ void k_pool_gate(const float* __restrict__ x, const float* __restrict__ gw,
                            const float* __restrict__ gb, const int* __restrict__ batch,
                            float* __restrict__ gate, float* __restrict__ gmax) {
  int wv = threadIdx.x >> 5, lane = threadIdx.x & 31;
  int n = blockIdx.x * 8 + wv;
  if (n >= NN) return;
  const float* xr = x + (size_t)n * HC;
  float s = 0.f;
#pragma unroll
  for (int j = lane; j < HC; j += 32) s += xr[j] * gw[j];
#pragma unroll
  for (int off = 16; off > 0; off >>= 1) s += __shfl_xor(s, off, 32);
  if (lane == 0) {
    float g = 1.f / (1.f + __expf(-(s + gb[0])));
    gate[n] = g;
    atomicMaxF(&gmax[batch[n]], g);
  }
}

__global__ void k_pool_gate_exp(const float* __restrict__ gate, const int* __restrict__ batch,
                                const float* __restrict__ gmax, float* __restrict__ gex,
                                float* __restrict__ gsum) {
  int i = blockIdx.x * blockDim.x + threadIdx.x;
  if (i >= NN) return;
  int b = batch[i];
  float ex = __expf(gate[i] - gmax[b]);
  gex[i] = ex;
  atomicAdd(&gsum[b], ex);
}

__global__ void k_pool_accum(const float* __restrict__ x, const int* __restrict__ batch,
                             const float* __restrict__ gex, const float* __restrict__ gsum,
                             float* __restrict__ patt, float* __restrict__ psum,
                             float* __restrict__ pmax) {
  int t = blockIdx.x * blockDim.x + threadIdx.x; // over NN*HC
  if (t >= NN * HC) return;
  int f = t & (HC - 1);
  int n = t >> 7;
  int b = batch[n];
  float xv = x[t];
  float a = gex[n] / (gsum[b] + 1e-16f);
  atomicAdd(&patt[b * HC + f], a * xv);
  atomicAdd(&psum[b * HC + f], xv);
  atomicMaxF(&pmax[b * HC + f], xv);
}

__global__ void k_pool_combine(const float* __restrict__ patt, const float* __restrict__ psum,
                               const float* __restrict__ pmax, const float* __restrict__ counts,
                               const float* __restrict__ pw, float* __restrict__ p) {
  int t = blockIdx.x * blockDim.x + threadIdx.x; // over GG*HC
  if (t >= GG * HC) return;
  int g = t >> 7;
  p[t] = pw[0] * patt[t] + pw[1] * psum[t] / counts[g] + pw[2] * pmax[t];
}

__global__ void k_pool_out(const float* __restrict__ p, const float* __restrict__ linW,
                           const float* __restrict__ linb, const float* __restrict__ hw,
                           float* __restrict__ oacc) {
  int t = blockIdx.x * blockDim.x + threadIdx.x; // over GG*OUTD
  if (t >= GG * OUTD) return;
  int g = t / OUTD, o = t - g * OUTD;
  float s = linb[o];
#pragma unroll 4
  for (int f = 0; f < HC; ++f) s += p[g * HC + f] * linW[f * OUTD + o];
  oacc[t] += hw[0] * s;
}

__global__ void k_risk(const float* __restrict__ oacc, const float* __restrict__ beta,
                       const float* __restrict__ h0, float* __restrict__ out) {
  int g = threadIdx.x;
  if (g < GG) {
    float s = h0[0];
#pragma unroll 4
    for (int o = 0; o < OUTD; ++o) s += oacc[g * OUTD + o] * beta[o];
    out[g] = s;
  }
}

// ---------------- launcher ----------------
extern "C" void kernel_launch(void* const* d_in, const int* in_sizes, int n_in,
                              void* d_out, int out_size, void* d_ws, size_t ws_size,
                              hipStream_t stream) {
  (void)in_sizes; (void)n_in; (void)out_size; (void)ws_size;
  const float* x_in   = (const float*)d_in[0];
  const int*   src    = (const int*)d_in[1];
  const int*   dst    = (const int*)d_in[2];
  const int*   batch  = (const int*)d_in[3];
  const float* Ws     = (const float*)d_in[4];
  const float* att_s  = (const float*)d_in[5];
  const float* att_d  = (const float*)d_in[6];
  const float* conv_b = (const float*)d_in[7];
  const float* gate_W = (const float*)d_in[8];
  const float* gate_b = (const float*)d_in[9];
  const float* lin_W  = (const float*)d_in[10];
  const float* lin_b  = (const float*)d_in[11];
  const float* hw     = (const float*)d_in[12];
  const float* h0     = (const float*)d_in[13];
  const float* beta   = (const float*)d_in[14];
  const float* pw     = (const float*)d_in[15];
  float* risk = (float*)d_out;

  // workspace carve (≈107 MB total)
  char* w = (char*)d_ws;
  auto carve = [&](size_t bytes) -> char* {
    char* p = w; w += (bytes + 255) & ~(size_t)255; return p;
  };
  float* xA    = (float*)carve((size_t)NN * HC * 4);
  float* xB    = (float*)carve((size_t)NN * HC * 4);
  float* hfeat = (float*)carve((size_t)NN * HC * 4);
  unsigned short* xb  = (unsigned short*)carve((size_t)NN * HC * 2);
  unsigned short* wtb = (unsigned short*)carve((size_t)HC * HC * 2);
  float* al_s  = (float*)carve((size_t)NN * HH * 4);
  float* al_d  = (float*)carve((size_t)NN * HH * 4);
  float* mmax  = (float*)carve((size_t)NN * HH * 4);
  float* ssum  = (float*)carve((size_t)NN * HH * 4);
  float* earr  = (float*)carve((size_t)ETOT * HH * 4);
  float* gate  = (float*)carve((size_t)NN * 4);
  float* gex   = (float*)carve((size_t)NN * 4);
  float* gmax  = (float*)carve(GG * 4);
  float* gsum  = (float*)carve(GG * 4);
  float* cnts  = (float*)carve(GG * 4);
  float* patt  = (float*)carve(GG * HC * 4);
  float* psum  = (float*)carve(GG * HC * 4);
  float* pmax  = (float*)carve(GG * HC * 4);
  float* pbuf  = (float*)carve(GG * HC * 4);
  float* oacc  = (float*)carve(GG * OUTD * 4);

  const int B = 256;
  auto g1 = [](int n) { return (n + 255) / 256; };
  const float NEGINF = -__builtin_inff();

  // one-time init
  k_fill<<<g1(GG), B, 0, stream>>>(cnts, 0.f, GG);
  k_counts<<<g1(NN), B, 0, stream>>>(batch, cnts);
  k_fill<<<g1(GG * OUTD), B, 0, stream>>>(oacc, 0.f, GG * OUTD);

  auto pool_block = [&](const float* x, int l) {
    k_fill<<<g1(GG), B, 0, stream>>>(gmax, NEGINF, GG);
    k_fill<<<g1(GG), B, 0, stream>>>(gsum, 0.f, GG);
    k_fill<<<g1(GG * HC), B, 0, stream>>>(patt, 0.f, GG * HC);
    k_fill<<<g1(GG * HC), B, 0, stream>>>(psum, 0.f, GG * HC);
    k_fill<<<g1(GG * HC), B, 0, stream>>>(pmax, NEGINF, GG * HC);
    k_pool_gate<<<(NN + 7) / 8, B, 0, stream>>>(x, gate_W + (size_t)l * HC, gate_b + l,
                                                batch, gate, gmax);
    k_pool_gate_exp<<<g1(NN), B, 0, stream>>>(gate, batch, gmax, gex, gsum);
    k_pool_accum<<<g1(NN * HC), B, 0, stream>>>(x, batch, gex, gsum, patt, psum, pmax);
    k_pool_combine<<<g1(GG * HC), B, 0, stream>>>(patt, psum, pmax, cnts, pw, pbuf);
    k_pool_out<<<g1(GG * OUTD), B, 0, stream>>>(pbuf, lin_W + (size_t)l * HC * OUTD,
                                                lin_b + (size_t)l * OUTD, hw + l, oacc);
  };

  const float* xcur = x_in;
  float* bufs[2] = {xA, xB};
  int pp = 0;

  for (int l = 0; l < LLAY; ++l) {
    pool_block(xcur, l);

    float* xnext = bufs[pp];
    // bf16 convert x and transposed W for the WMMA GEMM
    k_f32_to_bf16<<<g1(NN * HC), B, 0, stream>>>(xcur, xb, NN * HC);
    k_w_transpose_bf16<<<g1(HC * HC), B, 0, stream>>>(Ws + (size_t)l * HC * HC, wtb);
    k_gemm_wmma<<<NN / 16, B, 0, stream>>>(xb, wtb, conv_b + (size_t)l * HC, hfeat);
    k_attn_logits<<<g1(NN * HH), B, 0, stream>>>(hfeat, att_s + (size_t)l * HH * CC,
                                                 att_d + (size_t)l * HH * CC, al_s, al_d);
    k_fill<<<g1(NN * HH), B, 0, stream>>>(mmax, NEGINF, NN * HH);
    k_fill<<<g1(NN * HH), B, 0, stream>>>(ssum, 0.f, NN * HH);
    k_fill<<<g1(NN * HC), B, 0, stream>>>(xnext, 0.f, NN * HC);
    k_edge_max<<<g1(ETOT), B, 0, stream>>>(src, dst, al_s, al_d, earr, mmax);
    k_edge_expsum<<<g1(ETOT), B, 0, stream>>>(dst, earr, mmax, ssum);
    k_edge_scatter<<<g1(ETOT * HC), B, 0, stream>>>(src, dst, earr, ssum, hfeat, xnext);
    if (l != LLAY - 1) k_relu<<<g1(NN * HC), B, 0, stream>>>(xnext, NN * HC);

    xcur = xnext;
    pp ^= 1;
  }

  pool_block(xcur, LLAY);
  k_risk<<<1, 64, 0, stream>>>(oacc, beta, h0, risk);
}
